// SS2D_34059090657942
// MI455X (gfx1250) — compile-verified, matrix operand order checked by reading, not run on hardware
//
#include <hip/hip_runtime.h>
#include <hip/hip_bf16.h>
#include <math.h>

// ---------------- problem constants (from reference) ----------------
#define B_   4
#define H_   56
#define W_   56
#define DM   96            // d_model
#define DI   192           // d_inner
#define NS   16            // d_state
#define RK   6             // dt_rank
#define KG   4             // scan directions
#define L_   (H_ * W_)     // 3136
#define ML   (B_ * L_)     // 12544 rows for token-major GEMMs
#define CC   (RK + 2 * NS) // 38 projected channels
#define CP   40            // padded x_dbl row stride (16B aligned)

typedef __attribute__((ext_vector_type(16))) _Float16 v16h;
typedef __attribute__((ext_vector_type(8)))  float    v8f;

// ---------------- WMMA helpers (wave32, 16x16x32 f16 -> f32) ----------------
__device__ __forceinline__ v8f wmma_f16(v16h a, v16h b, v8f c) {
  // 8 args: (neg_a, A, neg_b, B, c_mod, C, reuse_a, reuse_b)
  return __builtin_amdgcn_wmma_f32_16x16x32_f16(false, a, false, b, (short)0, c,
                                                false, false);
}

// Build one 16x32 fragment (A or B) directly from global memory.
// ISA 7.12.2 layout: lane l -> (m|n) = l&15, half = l>>4; VGPR v packs
// K = (v/4)*16 + half*8 + (v%4)*2 .. +1.  Per lane that is two contiguous
// 8-float runs: [kc + half*8, +8) and [kc + 16 + half*8, +8).
__device__ __forceinline__ v16h frag_f32(const float* __restrict__ p0,
                                         const float* __restrict__ p1) {
  v16h f;
#pragma unroll
  for (int i = 0; i < 8; ++i) {
    f[i]     = (_Float16)p0[i];
    f[8 + i] = (_Float16)p1[i];
  }
  return f;
}

__device__ __forceinline__ v16h frag_f16(const _Float16* __restrict__ p0,
                                         const _Float16* __restrict__ p1) {
  v16h f;
#pragma unroll
  for (int i = 0; i < 8; ++i) {
    f[i]     = p0[i];
    f[8 + i] = p1[i];
  }
  return f;
}

__device__ __forceinline__ float silu_f(float x) {
  return x / (1.0f + expf(-x));
}

// scan-order position l -> image position (h*W+w) for direction k
__device__ __forceinline__ int scan_img(int k, int l) {
  switch (k & 3) {
    case 0:  return l;
    case 1:  return (l % H_) * W_ + (l / H_);
    case 2:  return L_ - 1 - l;
    default: { const int m = L_ - 1 - l; return (m % H_) * W_ + (m / H_); }
  }
}

// ---------------- kernel 1: in_proj GEMM + split + SiLU(z) ----------------
// xz[m, 0:384] = x[m, :] @ in_proj_w^T ; one wave = 16x64 output tile
// (A-fragment reused across 4 N-tiles).
__global__ __launch_bounds__(32) void k_inproj(const float* __restrict__ x,
                                               const float* __restrict__ w,
                                               float* __restrict__ xxt,
                                               float* __restrict__ zbuf) {
  const int NG = (2 * DI) / 64;                 // 6 groups of 4 N-tiles
  const int ng = blockIdx.x % NG;
  const int mt = blockIdx.x / NG;
  const int m0 = mt * 16, n0 = ng * 64;
  const int lane = threadIdx.x;
  const int mr = lane & 15, hs = lane >> 4;

  const float* __restrict__ arow = x + (size_t)(m0 + mr) * DM;
  v8f acc[4] = {};

#pragma unroll
  for (int kc = 0; kc < DM; kc += 32) {
    const v16h af = frag_f32(arow + kc + hs * 8, arow + kc + 16 + hs * 8);
#pragma unroll
    for (int j = 0; j < 4; ++j) {
      const float* __restrict__ brow = w + (size_t)(n0 + j * 16 + mr) * DM;
      const v16h bf = frag_f32(brow + kc + hs * 8, brow + kc + 16 + hs * 8);
      acc[j] = wmma_f16(af, bf, acc[j]);
    }
  }

#pragma unroll
  for (int j = 0; j < 4; ++j) {
#pragma unroll
    for (int v = 0; v < 8; ++v) {
      const int m = m0 + v + 8 * hs;
      const int n = n0 + j * 16 + mr;
      const float val = acc[j][v];
      if (n < DI) xxt[(size_t)m * DI + n] = val;
      else        zbuf[(size_t)m * DI + (n - DI)] = silu_f(val);
    }
  }
}

// ---------------- kernel 2: depthwise 3x3 conv + bias + SiLU ----------------
__global__ __launch_bounds__(256) void k_conv(const float* __restrict__ xxt,
                                              const float* __restrict__ cw,
                                              const float* __restrict__ cb,
                                              float* __restrict__ xct) {
  const int t = blockIdx.x * 256 + threadIdx.x;  // == (b*L + l)*DI + c
  const int c = t % DI;
  const int l = (t / DI) % L_;
  const int b = t / (DI * L_);
  const int h = l / W_, w = l % W_;
  float acc = cb[c];
#pragma unroll
  for (int i = 0; i < 3; ++i) {
    const int hh = h + i - 1;
    if (hh < 0 || hh >= H_) continue;
#pragma unroll
    for (int j = 0; j < 3; ++j) {
      const int ww = w + j - 1;
      if (ww < 0 || ww >= W_) continue;
      acc += cw[c * 9 + i * 3 + j] *
             xxt[((size_t)b * L_ + hh * W_ + ww) * DI + c];
    }
  }
  xct[(size_t)t] = silu_f(acc);
}

// ---------------- kernel 3: x_dbl = Wproj[k] (38x192) @ xs[b,k] (192xL) ------
// One wave = 48x16 output (38 valid rows, 3 M-tiles share the B-fragment).
// B columns are contiguous in d thanks to the [b,l,c] layout of xct.
// Output stored token-major [b,k,l,CP] so the scan reads one 160B column/step.
__global__ __launch_bounds__(32) void k_xdbl(const float* __restrict__ xct,
                                             const float* __restrict__ xpw,
                                             float* __restrict__ xdbl) {
  const int LT = L_ / 16;            // 196
  const int lt = blockIdx.x % LT;
  const int bk = blockIdx.x / LT;
  const int k = bk % KG, b = bk / KG;
  const int l0 = lt * 16;
  const int lane = threadIdx.x;
  const int mr = lane & 15, hs = lane >> 4;

  const int ig = scan_img(k, l0 + mr);
  const float* __restrict__ bcol = xct + ((size_t)b * L_ + ig) * DI;
  const float* __restrict__ a0 = xpw + ((size_t)k * CC + mr) * DI;
  const float* __restrict__ a1 = xpw + ((size_t)k * CC + 16 + mr) * DI;
  const float* __restrict__ a2 = xpw + ((size_t)k * CC + 32 + mr) * DI;

  v8f acc0 = {}, acc1 = {}, acc2 = {};
#pragma unroll
  for (int kc = 0; kc < DI; kc += 32) {
    const v16h bf  = frag_f32(bcol + kc + hs * 8, bcol + kc + 16 + hs * 8);
    const v16h af0 = frag_f32(a0 + kc + hs * 8, a0 + kc + 16 + hs * 8);
    const v16h af1 = frag_f32(a1 + kc + hs * 8, a1 + kc + 16 + hs * 8);
    v16h af2 = {};
    if (32 + mr < CC)  // rows 38..47 are zero padding
      af2 = frag_f32(a2 + kc + hs * 8, a2 + kc + 16 + hs * 8);
    acc0 = wmma_f16(af0, bf, acc0);
    acc1 = wmma_f16(af1, bf, acc1);
    acc2 = wmma_f16(af2, bf, acc2);
  }

  float* __restrict__ o =
      xdbl + (((size_t)b * KG + k) * L_ + l0 + mr) * CP;  // token-major
#pragma unroll
  for (int v = 0; v < 8; ++v) {
    const int c = v + 8 * hs;
    o[c]      = acc0[v];
    o[c + 16] = acc1[v];
    if (c + 32 < CC) o[c + 32] = acc2[v];
  }
}

// ---------------- kernel 4: selective scan ----------------
// one lane per (b,k,d); the 38-wide dt/B/C column is 10 float4 broadcast
// loads per step, u is d-coalesced, y store is d-coalesced.
__global__ __launch_bounds__(DI) void k_scan(const float* __restrict__ xct,
                                             const float* __restrict__ xdbl,
                                             const float* __restrict__ dtw,
                                             const float* __restrict__ dtb,
                                             const float* __restrict__ alogs,
                                             const float* __restrict__ dsv,
                                             float* __restrict__ outy) {
  const int k = blockIdx.x % KG, b = blockIdx.x / KG;
  const int d = threadIdx.x;

  float wdt[RK];
#pragma unroll
  for (int r = 0; r < RK; ++r) wdt[r] = dtw[((size_t)k * DI + d) * RK + r];
  const float bias = dtb[k * DI + d];
  float Arow[NS];
#pragma unroll
  for (int n = 0; n < NS; ++n)
    Arow[n] = -expf(alogs[((size_t)k * DI + d) * NS + n]);
  const float Dv = dsv[k * DI + d];

  float h[NS];
#pragma unroll
  for (int n = 0; n < NS; ++n) h[n] = 0.0f;

  const float* __restrict__ xd = xdbl + ((size_t)b * KG + k) * (size_t)L_ * CP;
  float* __restrict__ yo = outy + ((size_t)b * KG + k) * (size_t)L_ * DI;

  for (int l = 0; l < L_; ++l) {
    // prefetch upcoming dt/B/C column (gfx1250 global_prefetch_b8)
    __builtin_prefetch(xd + (size_t)(l + 8) * CP, 0, 1);

    const float4* __restrict__ c4 = (const float4*)(xd + (size_t)l * CP);
    float colv[CP];
#pragma unroll
    for (int i = 0; i < CP / 4; ++i) {
      const float4 t = c4[i];
      colv[4 * i + 0] = t.x;
      colv[4 * i + 1] = t.y;
      colv[4 * i + 2] = t.z;
      colv[4 * i + 3] = t.w;
    }

    float dr = bias;
#pragma unroll
    for (int r = 0; r < RK; ++r) dr += wdt[r] * colv[r];
    const float delta = (dr > 20.0f) ? dr : log1pf(expf(dr));

    const int ig = scan_img(k, l);
    const float u = xct[((size_t)b * L_ + ig) * DI + d];
    const float du = delta * u;

    float y = 0.0f;
#pragma unroll
    for (int n = 0; n < NS; ++n) {
      h[n] = expf(delta * Arow[n]) * h[n] + du * colv[RK + n];
      y += h[n] * colv[RK + NS + n];
    }
    yo[(size_t)l * DI + d] = y + Dv * u;   // [b,k,l,d] -> coalesced
  }
}

// ---------------- kernel 5: cross-merge + LayerNorm + gate, emit f16 --------
__global__ __launch_bounds__(256) void k_lngate(const float* __restrict__ outy,
                                                const float* __restrict__ zbuf,
                                                const float* __restrict__ lnw,
                                                const float* __restrict__ lnb,
                                                _Float16* __restrict__ yg) {
  const int row = blockIdx.x * 8 + (threadIdx.x >> 5);   // m in [0, ML)
  const int lane = threadIdx.x & 31;
  const int b = row / L_, l = row % L_;

  int pos[KG];
  pos[0] = l;
  pos[1] = (l % W_) * H_ + (l / W_);
  pos[2] = L_ - 1 - l;
  pos[3] = L_ - 1 - pos[1];

  float vals[DI / 32];
  float s1 = 0.0f, s2 = 0.0f;
#pragma unroll
  for (int j = 0; j < DI / 32; ++j) {
    const int c = lane + 32 * j;
    float v = 0.0f;
#pragma unroll
    for (int k = 0; k < KG; ++k)
      v += outy[(((size_t)b * KG + k) * L_ + pos[k]) * DI + c];
    vals[j] = v;
    s1 += v;
    s2 += v * v;
  }
#pragma unroll
  for (int o = 16; o > 0; o >>= 1) {
    s1 += __shfl_xor(s1, o, 32);
    s2 += __shfl_xor(s2, o, 32);
  }
  const float mu = s1 / (float)DI;
  const float rstd = rsqrtf(s2 / (float)DI - mu * mu + 1e-5f);
#pragma unroll
  for (int j = 0; j < DI / 32; ++j) {
    const int c = lane + 32 * j;
    float v = (vals[j] - mu) * rstd * lnw[c] + lnb[c];
    v *= zbuf[(size_t)row * DI + c];
    yg[(size_t)row * DI + c] = (_Float16)v;
  }
}

// ---------------- kernel 6: out_proj GEMM -> d_out ----------------
// One wave = 16x96 output row-strip: A-fragment (f16) reused across all
// 6 N-tiles; 36 WMMAs per wave.
__global__ __launch_bounds__(32) void k_outproj(const _Float16* __restrict__ yg,
                                                const float* __restrict__ wo,
                                                float* __restrict__ out) {
  const int m0 = blockIdx.x * 16;
  const int lane = threadIdx.x;
  const int mr = lane & 15, hs = lane >> 4;

  const _Float16* __restrict__ arow = yg + (size_t)(m0 + mr) * DI;
  v8f acc[DM / 16] = {};

#pragma unroll
  for (int kc = 0; kc < DI; kc += 32) {
    const v16h af = frag_f16(arow + kc + hs * 8, arow + kc + 16 + hs * 8);
#pragma unroll
    for (int j = 0; j < DM / 16; ++j) {
      const float* __restrict__ brow = wo + (size_t)(j * 16 + mr) * DI;
      const v16h bf = frag_f32(brow + kc + hs * 8, brow + kc + 16 + hs * 8);
      acc[j] = wmma_f16(af, bf, acc[j]);
    }
  }

#pragma unroll
  for (int j = 0; j < DM / 16; ++j)
#pragma unroll
    for (int v = 0; v < 8; ++v)
      out[(size_t)(m0 + v + 8 * hs) * DM + j * 16 + mr] = acc[j][v];
}

// ---------------- host launcher ----------------
extern "C" void kernel_launch(void* const* d_in, const int* in_sizes, int n_in,
                              void* d_out, int out_size, void* d_ws, size_t ws_size,
                              hipStream_t stream) {
  const float* x    = (const float*)d_in[0];   // [B,H,W,DM]
  const float* ipw  = (const float*)d_in[1];   // [2*DI, DM]
  const float* cw   = (const float*)d_in[2];   // [DI,1,3,3]
  const float* cb   = (const float*)d_in[3];   // [DI]
  const float* xpw  = (const float*)d_in[4];   // [KG, 38, DI]
  const float* dtw  = (const float*)d_in[5];   // [KG, DI, RK]
  const float* dtb  = (const float*)d_in[6];   // [KG, DI]
  const float* alog = (const float*)d_in[7];   // [KG*DI, NS]
  const float* dsv  = (const float*)d_in[8];   // [KG*DI]
  const float* lnw  = (const float*)d_in[9];   // [DI]
  const float* lnb  = (const float*)d_in[10];  // [DI]
  const float* wo   = (const float*)d_in[11];  // [DM, DI]
  float* out = (float*)d_out;                  // [B,H,W,DM]

  // workspace carve-up (~80 MB)
  float* xxt  = (float*)d_ws;                          // [ML, DI]
  float* zbuf = xxt + (size_t)ML * DI;                 // [ML, DI]
  float* xct  = zbuf + (size_t)ML * DI;                // [ML, DI]
  float* xdbl = xct + (size_t)ML * DI;                 // [B,K,L,CP]
  float* outy = xdbl + (size_t)B_ * KG * L_ * CP;      // [B,K,L,DI]
  _Float16* yg = (_Float16*)(outy + (size_t)B_ * KG * L_ * DI); // [ML, DI]

  (void)in_sizes; (void)n_in; (void)out_size; (void)ws_size;

  k_inproj <<<(ML / 16) * ((2 * DI) / 64), 32, 0, stream>>>(x, ipw, xxt, zbuf);
  k_conv   <<<(B_ * L_ * DI) / 256, 256, 0, stream>>>(xxt, cw, cb, xct);
  k_xdbl   <<<B_ * KG * (L_ / 16), 32, 0, stream>>>(xct, xpw, xdbl);
  k_scan   <<<B_ * KG, DI, 0, stream>>>(xct, xdbl, dtw, dtb, alog, dsv, outy);
  k_lngate <<<ML / 8, 256, 0, stream>>>(outy, zbuf, lnw, lnb, yg);
  k_outproj<<<ML / 16, 32, 0, stream>>>(yg, wo, out);
}
